// RefineNet_shallow_42923903156751
// MI455X (gfx1250) — matrix operand order repaired and checked by reading, not compiled, not run on hardware
//
#include <hip/hip_runtime.h>
#include <hip/hip_bf16.h>
#include <hip/hip_fp16.h>

#define B_  8
#define H_  192
#define W_  640
#define Z_  16
#define N_  (B_*H_*W_)          // 983040, divisible by 256 and 16
#define EPS_ 1e-5f

typedef __attribute__((ext_vector_type(16))) _Float16 v16h;
typedef __attribute__((ext_vector_type(8)))  float    v8f;

// ---------------------------------------------------------------------------
// Kernel 0: per-point neighbor code.  For each of the 9 spatial offsets
// (dx,dy), store 2 bits: dz+1 in {0,1,2} if a z-matching neighbor exists,
// else 3.  Also zero the pad rows (row N) of both activation buffers.
// ---------------------------------------------------------------------------
__global__ void k_nbrcode(const int* __restrict__ coords, int* __restrict__ nbrcode,
                          _Float16* __restrict__ x1pad, _Float16* __restrict__ x2pad) {
    int i = blockIdx.x * blockDim.x + threadIdx.x;
    if (i >= N_) return;
    int rem = i % (H_ * W_);
    int v = rem / W_, u = rem % W_;
    int zi = coords[4*i + 3];
    unsigned code = 0;
#pragma unroll
    for (int off9 = 0; off9 < 9; ++off9) {
        int dx = off9 / 3 - 1, dy = off9 % 3 - 1;
        unsigned c = 3u;
        int uu = u + dx, vv = v + dy;
        if (uu >= 0 && uu < W_ && vv >= 0 && vv < H_) {
            int j  = i + dy * W_ + dx;
            int dz = coords[4*j + 3] - zi;
            if (dz >= -1 && dz <= 1) c = (unsigned)(dz + 1);
        }
        code |= c << (2 * off9);
    }
    nbrcode[i] = (int)code;
    if (i == 0) {
#pragma unroll
        for (int c2 = 0; c2 < 16; ++c2) {
            x1pad[N_*16 + c2] = (_Float16)0.f;
            x2pad[N_*16 + c2] = (_Float16)0.f;
        }
    }
}

// ---------------------------------------------------------------------------
// Kernel 0b: pre-swizzle W2 into the WMMA B-operand per-lane stream.
// Pair p covers taps k0=2p (dwords 0-3) and k1=2p+1 (dwords 4-7, zero if 27).
// dword d of lane (h=lane>>4, n=lane&15) = f16 pair (W2[k][c][n], W2[k][c+1][n]), k=2p+(d>>2), c=8h+2(d&3).
// ---------------------------------------------------------------------------
__global__ void k_packW2(const float* __restrict__ W2, unsigned* __restrict__ B2) {
    int t = blockIdx.x * blockDim.x + threadIdx.x;
    if (t >= 14 * 32 * 8) return;
    int d    = t & 7;
    int lane = (t >> 3) & 31;
    int p    = t >> 8;
    int h = lane >> 4, n = lane & 15;
    int k = 2 * p + (d >> 2);
    unsigned out = 0;
    if (k < 27) {
        _Float16 lo = (_Float16)W2[k*256 + (8*h + 2*(d&3)    )*16 + n];
        _Float16 hi = (_Float16)W2[k*256 + (8*h + 2*(d&3) + 1)*16 + n];
        out = (unsigned)__builtin_bit_cast(unsigned short, lo)
            | ((unsigned)__builtin_bit_cast(unsigned short, hi) << 16);
    }
    B2[t] = out;
}

// ---------------------------------------------------------------------------
// Layer 1: 5 -> 16 channels, VALU FMA (too skinny for WMMA), fused BN+ReLU,
// output written as f16 (WMMA A operand precision for layer 2).
// ---------------------------------------------------------------------------
__global__ void k_layer1(const float* __restrict__ feats, const int* __restrict__ nbrcode,
                         const float* __restrict__ W1,
                         const float* __restrict__ g1, const float* __restrict__ be1,
                         const float* __restrict__ m1, const float* __restrict__ v1,
                         _Float16* __restrict__ x1pad) {
    __shared__ float sW[27*5*16];
    __shared__ float sbn[48];
    for (int t = threadIdx.x; t < 27*5*16; t += blockDim.x) sW[t] = W1[t];
    if (threadIdx.x < 16) {
        sbn[threadIdx.x]      = g1[threadIdx.x] * rsqrtf(v1[threadIdx.x] + EPS_);
        sbn[16 + threadIdx.x] = m1[threadIdx.x];
        sbn[32 + threadIdx.x] = be1[threadIdx.x];
    }
    __syncthreads();
    int i = blockIdx.x * blockDim.x + threadIdx.x;
    if (i >= N_) return;
    unsigned code = (unsigned)nbrcode[i];
    float acc[16];
#pragma unroll
    for (int c = 0; c < 16; ++c) acc[c] = 0.f;
#pragma unroll
    for (int off9 = 0; off9 < 9; ++off9) {
        unsigned c2 = (code >> (2*off9)) & 3u;
        if (c2 == 3u) continue;
        int dx = off9 / 3 - 1, dy = off9 % 3 - 1;
        int j = i + dy * W_ + dx;
        int k = off9 * 3 + (int)c2;
        const float* fp = feats + j * 5;
        float f0 = fp[0], f1 = fp[1], f2 = fp[2], f3 = fp[3], f4 = fp[4];
        const float* wp = sW + k * 80;
#pragma unroll
        for (int c = 0; c < 16; ++c)
            acc[c] += f0*wp[c] + f1*wp[16+c] + f2*wp[32+c] + f3*wp[48+c] + f4*wp[64+c];
    }
    unsigned up[8];
#pragma unroll
    for (int c = 0; c < 16; c += 2) {
        float y0 = fmaxf((acc[c]   - sbn[16+c])   * sbn[c]   + sbn[32+c],   0.f);
        float y1 = fmaxf((acc[c+1] - sbn[16+c+1]) * sbn[c+1] + sbn[32+c+1], 0.f);
        _Float16 h0 = (_Float16)y0, h1 = (_Float16)y1;
        up[c/2] = (unsigned)__builtin_bit_cast(unsigned short, h0)
                | ((unsigned)__builtin_bit_cast(unsigned short, h1) << 16);
    }
    uint4* dst = (uint4*)(x1pad + i * 16);
    dst[0] = make_uint4(up[0], up[1], up[2], up[3]);
    dst[1] = make_uint4(up[4], up[5], up[6], up[7]);
}

// ---------------------------------------------------------------------------
// Layer 2: 16 -> 16 channels via v_wmma_f32_16x16x32_f16.
// One wave = one 16-point tile; 14 WMMAs cover all 27 taps (2 taps / WMMA).
// A gathers are software-pipelined with a depth-2 prefetch so each WMMA waits
// on loads issued two iterations earlier instead of wait-to-zero.
// Fused BN+ReLU on the f32 accumulator; store f16.
// ---------------------------------------------------------------------------
__global__ void k_layer2(const _Float16* __restrict__ x1pad, const int* __restrict__ nbrcode,
                         const unsigned* __restrict__ B2,
                         const float* __restrict__ g2, const float* __restrict__ be2,
                         const float* __restrict__ m2, const float* __restrict__ v2,
                         _Float16* __restrict__ x2pad) {
    __shared__ unsigned sB[14 * 256];
    __shared__ float sbn[48];
    for (int t = threadIdx.x; t < 14 * 256; t += blockDim.x) sB[t] = B2[t];
    if (threadIdx.x < 16) {
        sbn[threadIdx.x]      = g2[threadIdx.x] * rsqrtf(v2[threadIdx.x] + EPS_);
        sbn[16 + threadIdx.x] = m2[threadIdx.x];
        sbn[32 + threadIdx.x] = be2[threadIdx.x];
    }
    __syncthreads();

    int lane = threadIdx.x & 31;
    int wave = threadIdx.x >> 5;
    int tile = blockIdx.x * 8 + wave;
    int base = tile * 16;
    int m = lane & 15;          // point within tile (A row / D column group)
    int h = lane >> 4;          // which channel half this lane carries
    int i = base + m;
    unsigned code = (unsigned)nbrcode[i];

    // Row index for every tap (pad row N_ when the tap is inactive).
    int jrow[28];
#pragma unroll
    for (int k = 0; k < 27; ++k) {
        const int off9 = k / 3, need = k % 3;
        const int dx = off9 / 3 - 1, dy = off9 % 3 - 1;
        jrow[k] = (((code >> (2*off9)) & 3u) == (unsigned)need) ? (i + dy*W_ + dx) : N_;
    }
    jrow[27] = N_;              // zero row pads tap 27

    // A operand gather for pair p: lane<16 carries K 0-7 (k0) + 16-23 (k1);
    // lane>=16 carries K 8-15 + 24-31 -> channel slice = h*8 of each tap row.
    const _Float16* xh = x1pad + h * 8;
#define LDA(dst, p)                                                      \
    do {                                                                 \
        (dst)[0] = *(const uint4*)(xh + jrow[2*(p)]     * 16);           \
        (dst)[1] = *(const uint4*)(xh + jrow[2*(p) + 1] * 16);           \
    } while (0)

    uint4 buf0[2], buf1[2], buf2[2];
    LDA(buf0, 0);
    LDA(buf1, 1);

    v8f acc = {};
#pragma unroll
    for (int p = 0; p < 14; ++p) {
        // depth-2 prefetch into the buffer slot that frees up this iteration
        uint4* cur  = (p % 3 == 0) ? buf0 : (p % 3 == 1) ? buf1 : buf2;
        uint4* next = (p % 3 == 0) ? buf2 : (p % 3 == 1) ? buf0 : buf1;
        if (p + 2 < 14) LDA(next, p + 2);

        union { v16h hh; uint4 q[2]; } A;
        A.q[0] = cur[0]; A.q[1] = cur[1];
        union { v16h hh; uint4 q[2]; } Bv;
        const uint4* bp = (const uint4*)(sB + p * 256 + lane * 8);
        Bv.q[0] = bp[0]; Bv.q[1] = bp[1];
        acc = __builtin_amdgcn_wmma_f32_16x16x32_f16(
                  false, A.hh, false, Bv.hh, (short)0, acc, false, false);
    }
#undef LDA

    // D layout: lane<16 -> rows r (M=r), lane>=16 -> rows r+8; column n = lane&15.
    const int n = m;
    const float scale = sbn[n], mean = sbn[16 + n], bias = sbn[32 + n];
#pragma unroll
    for (int r = 0; r < 8; ++r) {
        int row = r + 8 * h;
        float y = fmaxf((acc[r] - mean) * scale + bias, 0.f);
        x2pad[(base + row) * 16 + n] = (_Float16)y;
    }
}

// ---------------------------------------------------------------------------
// Layer 3: 16 -> 1 channel, VALU FMA, + b3, tanh, f32 output.
// ---------------------------------------------------------------------------
__global__ void k_layer3(const _Float16* __restrict__ x2pad, const int* __restrict__ nbrcode,
                         const float* __restrict__ W3, const float* __restrict__ b3,
                         float* __restrict__ out) {
    __shared__ float sW[27 * 16];
    for (int t = threadIdx.x; t < 27 * 16; t += blockDim.x) sW[t] = W3[t];
    __syncthreads();
    int i = blockIdx.x * blockDim.x + threadIdx.x;
    if (i >= N_) return;
    unsigned code = (unsigned)nbrcode[i];
    float acc = 0.f;
#pragma unroll
    for (int off9 = 0; off9 < 9; ++off9) {
        unsigned c2 = (code >> (2*off9)) & 3u;
        if (c2 == 3u) continue;
        int dx = off9 / 3 - 1, dy = off9 % 3 - 1;
        int j = i + dy * W_ + dx;
        const float*    wp = sW + (off9 * 3 + (int)c2) * 16;
        const _Float16* xp = x2pad + j * 16;
#pragma unroll
        for (int ci = 0; ci < 16; ++ci)
            acc += (float)xp[ci] * wp[ci];
    }
    out[i] = tanhf(acc + b3[0]);
}

// ---------------------------------------------------------------------------
// Host-side launcher.
// Input order: 0 feats, 1 coords, 2 W1, 3 g1, 4 be1, 5 m1, 6 v1,
//              7 W2, 8 g2, 9 be2, 10 m2, 11 v2, 12 W3, 13 b3, 14 batch_size
// ---------------------------------------------------------------------------
extern "C" void kernel_launch(void* const* d_in, const int* in_sizes, int n_in,
                              void* d_out, int out_size, void* d_ws, size_t ws_size,
                              hipStream_t stream) {
    const float* feats  = (const float*)d_in[0];
    const int*   coords = (const int*)  d_in[1];
    const float* W1  = (const float*)d_in[2];
    const float* g1  = (const float*)d_in[3];
    const float* be1 = (const float*)d_in[4];
    const float* m1  = (const float*)d_in[5];
    const float* v1  = (const float*)d_in[6];
    const float* W2  = (const float*)d_in[7];
    const float* g2  = (const float*)d_in[8];
    const float* be2 = (const float*)d_in[9];
    const float* m2  = (const float*)d_in[10];
    const float* v2  = (const float*)d_in[11];
    const float* W3  = (const float*)d_in[12];
    const float* b3  = (const float*)d_in[13];
    float* out = (float*)d_out;

    // workspace layout (all 16B aligned)
    char* ws = (char*)d_ws;
    const size_t OFF_CODE = 0;
    const size_t SZ_CODE  = (size_t)N_ * 4;                     //  3,932,160
    const size_t OFF_X1   = OFF_CODE + SZ_CODE;
    const size_t SZ_X     = (size_t)(N_ + 1) * 16 * 2;          // 31,457,312
    const size_t OFF_X2   = OFF_X1 + SZ_X;
    const size_t OFF_B2   = OFF_X2 + SZ_X;                      // +14,336 -> ~63.8 MB total

    int*      nbrcode = (int*)     (ws + OFF_CODE);
    _Float16* x1pad   = (_Float16*)(ws + OFF_X1);
    _Float16* x2pad   = (_Float16*)(ws + OFF_X2);
    unsigned* B2      = (unsigned*)(ws + OFF_B2);

    const int TPB = 256;
    const int NBLK = N_ / TPB;          // 3840

    k_nbrcode<<<NBLK, TPB, 0, stream>>>(coords, nbrcode, x1pad, x2pad);
    k_packW2 <<<14,   TPB, 0, stream>>>(W2, B2);
    k_layer1 <<<NBLK, TPB, 0, stream>>>(feats, nbrcode, W1, g1, be1, m1, v1, x1pad);
    k_layer2 <<<(N_/16)/8, TPB, 0, stream>>>(x1pad, nbrcode, B2, g2, be2, m2, v2, x2pad);
    k_layer3 <<<NBLK, TPB, 0, stream>>>(x2pad, nbrcode, W3, b3, out);
}
